// YoloV3_79680233276373
// MI455X (gfx1250) — matrix operand order repaired and checked by reading, not compiled, not run on hardware
//
#include <hip/hip_runtime.h>
#include <hip/hip_bf16.h>

typedef __attribute__((ext_vector_type(16))) _Float16 v16h;
typedef __attribute__((ext_vector_type(8)))  float    v8f;
typedef __attribute__((ext_vector_type(4)))  unsigned int u32x4;
typedef __attribute__((ext_vector_type(8)))  int i32x8;
typedef __attribute__((ext_vector_type(4)))  int i32x4;

#define BM 128
#define BN 64
#define BK 32

#if defined(__has_builtin)
#if __has_builtin(__builtin_amdgcn_tensor_load_to_lds) && __has_builtin(__builtin_amdgcn_s_wait_tensorcnt)
#define HAVE_TDM 1
#endif
#endif

__constant__ float c_anchors[18] = {
  10.f/416.f, 13.f/416.f,  16.f/416.f, 30.f/416.f,  33.f/416.f, 23.f/416.f,
  30.f/416.f, 61.f/416.f,  62.f/416.f, 45.f/416.f,  59.f/416.f, 119.f/416.f,
  116.f/416.f, 90.f/416.f, 156.f/416.f, 198.f/416.f, 373.f/416.f, 326.f/416.f };

#ifdef HAVE_TDM
// ---------------------------------------------------------------------------
// TDM: DMA a 2D f32 tile (32 rows x 64 cols, row stride = `stride` elems)
// from global into LDS at byte offset ldsOff, with LDS padding of 4 dwords
// every 64 dwords (-> row stride 68 floats). OOB rows/cols are zero-filled
// by the TDM per the D# tensor dims. Issued by one wave; EXEC is ignored.
// ---------------------------------------------------------------------------
static __device__ inline void tdm_load_b32_tile(const float* gaddr, unsigned ldsOff,
                                                int width, int rows, int stride)
{
  unsigned long long ga = (unsigned long long)(size_t)gaddr;
  u32x4 g0;
  g0[0] = 1u;                                        // count=1, user descriptor
  g0[1] = ldsOff;                                    // lds_addr (bytes)
  g0[2] = (unsigned)(ga & 0xffffffffull);            // global_addr[31:0]
  g0[3] = (unsigned)((ga >> 32) & 0x01ffffffull)     // global_addr[56:32]
        | (2u << 30);                                // type = 2 ("image")
  i32x8 g1;
  g1[0] = (int)((2u << 16)                           // data_size = 4B
              | (1u << 20)                           // pad_enable
              | (5u << 22)                           // pad_interval: 64 dwords
              | (3u << 25));                         // pad_amount: 4 dwords
  g1[1] = (int)(((unsigned)width & 0xffffu) << 16);  // tensor_dim0[15:0] @bits63:48
  g1[2] = (int)((((unsigned)width >> 16) & 0xffffu)  // tensor_dim0[31:16]
              | (((unsigned)rows & 0xffffu) << 16)); // tensor_dim1[15:0]
  g1[3] = (int)((((unsigned)rows >> 16) & 0xffffu)   // tensor_dim1[31:16]
              | (64u << 16));                        // tile_dim0 = 64
  g1[4] = (int)32u;                                  // tile_dim1 = 32, tile_dim2 = 0
  g1[5] = (int)stride;                               // tensor_dim0_stride[31:0]
  g1[6] = 0;                                         // stride hi / dim1_stride lo
  g1[7] = 0;
  i32x4 g2 = {0, 0, 0, 0};
  i32x4 g3 = {0, 0, 0, 0};
#if __clang_major__ >= 23
  i32x8 g4 = {0, 0, 0, 0, 0, 0, 0, 0};
  __builtin_amdgcn_tensor_load_to_lds(g0, g1, g2, g3, g4, 0);
#else
  __builtin_amdgcn_tensor_load_to_lds(g0, g1, g2, g3, 0);
#endif
}
#endif

// ---------------------------------------------------------------------------
// Implicit-GEMM convolution, NHWC x HWIO -> NHWC, f16 WMMA with f32 accum.
// K loop is (ky, kx, ci-chunk) so there is NO integer div/mod in the hot
// loop; A-tile staging is contiguous (vectorizes to global_load_b128) and
// B-tile staging is a TDM DMA. Fused BN + LeakyReLU(0.1) or bias, plus
// optional residual add after activation (Darknet residual semantics).
// ---------------------------------------------------------------------------
__global__ __launch_bounds__(256)
void conv_wmma_kernel(const float* __restrict__ in, const float* __restrict__ wgt,
                      const float* __restrict__ gamma, const float* __restrict__ beta,
                      const float* __restrict__ mean,  const float* __restrict__ var,
                      const float* __restrict__ bias,  const float* __restrict__ resid,
                      float* __restrict__ out,
                      int Nb, int H, int W, int Cin, int Cout,
                      int KH, int stride, int pad, int OH, int OW)
{
  __shared__ _Float16 As[BM][BK + 8];   // halves, padded row stride
  __shared__ float    Bsf[BK][BN + 4];  // f32, stride 68 matches TDM padding

  const int tid  = threadIdx.x;
  const int lane = tid & 31;
  const int wave = tid >> 5;      // 8 waves
  const int wm   = wave >> 1;     // 0..3 (M direction, 32 rows each)
  const int wn   = wave & 1;      // 0..1 (N direction, 32 cols each)

  const int M  = Nb * OH * OW;
  const int m0 = (int)blockIdx.x * BM;
  const int n0 = (int)blockIdx.y * BN;

  // A staging assignment: each thread fills one half-row of 16 halves.
  const int a_row = tid >> 1;
  const int a_kb  = (tid & 1) << 4;
  int a_n = 0, a_oh = 0, a_ow = 0;
  bool a_valid = false;
  {
    int m = m0 + a_row;
    if (m < M) {
      a_valid = true;
      a_ow = m % OW; int t = m / OW;
      a_oh = t % OH; a_n  = t / OH;
    }
  }
  // B fallback staging assignment (only used without TDM).
  const int b_r  = tid >> 3;
  const int b_cb = (tid & 7) << 3;

  v8f acc[2][2];
  #pragma unroll
  for (int i = 0; i < 2; ++i)
    #pragma unroll
    for (int j = 0; j < 2; ++j)
      #pragma unroll
      for (int e = 0; e < 8; ++e) acc[i][j][e] = 0.f;

  const int hl = lane >> 4;       // lane half (0/1)
  const int ll = lane & 15;

  for (int ky = 0; ky < KH; ++ky) {
    const int ih = a_oh * stride - pad + ky;
    for (int kx = 0; kx < KH; ++kx) {
      const int iw = a_ow * stride - pad + kx;
      const bool rowok = a_valid && ih >= 0 && ih < H && iw >= 0 && iw < W;
      const float* arow = in + (((size_t)a_n * H + ih) * W + iw) * Cin;
      const int kbase = (ky * KH + kx) * Cin;

      for (int c0 = 0; c0 < Cin; c0 += BK) {
        // ---- stage B tile: TDM DMA (one wave) or manual fallback ----
#ifdef HAVE_TDM
        if (wave == 0) {
          int rows = Cin - c0; if (rows > BK) rows = BK;
          tdm_load_b32_tile(wgt + ((size_t)(kbase + c0) * Cout + n0),
                            (unsigned)(size_t)&Bsf[0][0],
                            Cout - n0, rows, Cout);
        }
#else
        {
          int kk = kbase + c0 + b_r;
          bool rok = (c0 + b_r) < Cin;
          #pragma unroll
          for (int j = 0; j < 8; ++j) {
            int co = n0 + b_cb + j;
            float v = 0.f;
            if (rok && co < Cout) v = wgt[(size_t)kk * Cout + co];
            Bsf[b_r][b_cb + j] = v;
          }
        }
#endif
        // ---- stage A tile (contiguous f32 -> f16) ----
        {
          int cb = c0 + a_kb;
          if (rowok && (cb + 16) <= Cin) {
            #pragma unroll
            for (int j = 0; j < 16; ++j)
              As[a_row][a_kb + j] = (_Float16)arow[cb + j];
            if (cb + BK < Cin)       // hint next slab (global_prefetch_b8)
              __builtin_prefetch(arow + cb + BK, 0, 1);
          } else {
            #pragma unroll
            for (int j = 0; j < 16; ++j) {
              int ci = cb + j;
              float v = 0.f;
              if (rowok && ci < Cin) v = arow[ci];
              As[a_row][a_kb + j] = (_Float16)v;
            }
          }
        }
#ifdef HAVE_TDM
        if (wave == 0) __builtin_amdgcn_s_wait_tensorcnt(0);
#endif
        __syncthreads();

        // ---- fragments per ISA 7.12.2 layouts ----
        union AV { v16h v; unsigned int u[8]; };
        union BV { v16h v; _Float16 h[16]; };
        AV afrag[2]; BV bfrag[2];

        #pragma unroll
        for (int mi = 0; mi < 2; ++mi) {
          int row = wm * 32 + mi * 16 + ll;
          #pragma unroll
          for (int v = 0; v < 8; ++v) {
            // A 16x32 f16: VGPR v holds K = (v>=4?16:0) + lanehalf*8 + (v%4)*2 (+1)
            int k = ((v & 4) << 2) + (hl << 3) + ((v & 3) << 1);
            afrag[mi].u[v] = *(const unsigned int*)&As[row][k];
          }
        }
        #pragma unroll
        for (int ni = 0; ni < 2; ++ni) {
          int col = wn * 32 + ni * 16 + ll;
          #pragma unroll
          for (int v = 0; v < 8; ++v) {
            // B 32x16 f16: VGPR v holds K = (v>=4?16:0) + lanehalf*8 + (v%4)*2 (+1), N = lane%16
            int k = ((v & 4) << 2) + (hl << 3) + ((v & 3) << 1);
            bfrag[ni].h[2 * v]     = (_Float16)Bsf[k][col];
            bfrag[ni].h[2 * v + 1] = (_Float16)Bsf[k + 1][col];
          }
        }

        #pragma unroll
        for (int mi = 0; mi < 2; ++mi)
          #pragma unroll
          for (int ni = 0; ni < 2; ++ni)
            acc[mi][ni] = __builtin_amdgcn_wmma_f32_16x16x32_f16(
                false, afrag[mi].v, false, bfrag[ni].v,
                (short)0, acc[mi][ni], false, false);

        __syncthreads();
      }
    }
  }

  // ---- epilogue: BN + LeakyReLU (or bias) + optional residual ----
  #pragma unroll
  for (int mi = 0; mi < 2; ++mi) {
    #pragma unroll
    for (int ni = 0; ni < 2; ++ni) {
      int gn = n0 + wn * 32 + ni * 16 + ll;
      if (gn >= Cout) continue;
      float s = 1.f, sh = 0.f; bool act = false;
      if (gamma) {
        float iv = rsqrtf(var[gn] + 1e-3f);
        s = gamma[gn] * iv;
        sh = beta[gn] - mean[gn] * s;
        act = true;
      } else if (bias) {
        sh = bias[gn];
      }
      #pragma unroll
      for (int r = 0; r < 8; ++r) {
        int gm = m0 + wm * 32 + mi * 16 + hl * 8 + r;  // C/D: row = 8*lanehalf + vgpr
        if (gm >= M) continue;
        float y = acc[mi][ni][r] * s + sh;
        if (act) y = (y > 0.f) ? y : 0.1f * y;
        size_t o = (size_t)gm * Cout + gn;
        if (resid) y += resid[o];
        out[o] = y;
      }
    }
  }
}

// ---------------------------------------------------------------------------
// Elementwise helpers
// ---------------------------------------------------------------------------
__global__ void upsample2x_kernel(const float* __restrict__ in, float* __restrict__ out,
                                  int Nb, int H, int W, int C)
{
  size_t total = (size_t)Nb * 2 * H * 2 * W * C;
  size_t i = (size_t)blockIdx.x * blockDim.x + threadIdx.x;
  if (i >= total) return;
  int c = (int)(i % C); size_t t = i / C;
  int ow = (int)(t % (2 * W)); t /= (2 * W);
  int oh = (int)(t % (2 * H)); int n = (int)(t / (2 * H));
  out[i] = in[(((size_t)n * H + (oh >> 1)) * W + (ow >> 1)) * C + c];
}

__global__ void concat_ch_kernel(const float* __restrict__ a, const float* __restrict__ b,
                                 float* __restrict__ out, int Nb, int H, int W, int Ca, int Cb)
{
  int C = Ca + Cb;
  size_t total = (size_t)Nb * H * W * C;
  size_t i = (size_t)blockIdx.x * blockDim.x + threadIdx.x;
  if (i >= total) return;
  int c = (int)(i % C); size_t pix = i / C;
  out[i] = (c < Ca) ? a[pix * Ca + c] : b[pix * Cb + (c - Ca)];
}

// ---------------------------------------------------------------------------
// YOLO decode: sigmoid xy/obj/cls, exp wh * anchors, grid offset, corner boxes
// ---------------------------------------------------------------------------
__global__ void decode_kernel(const float* __restrict__ pred, float* __restrict__ boxes,
                              float* __restrict__ scores, int Nb, int g,
                              int anchor_off, int box_base, int total)
{
  int per = g * g * 3;
  int idx = blockIdx.x * blockDim.x + threadIdx.x;
  if (idx >= Nb * per) return;
  int n = idx / per, r = idx % per;
  int a = r % 3, cell = r / 3;
  int gx = cell % g, gy = cell / g;
  const float* p = pred + ((size_t)n * g * g + cell) * 18 + a * 6;
  float sx = 1.f / (1.f + __expf(-p[0]));
  float sy = 1.f / (1.f + __expf(-p[1]));
  float bw = __expf(p[2]) * c_anchors[(anchor_off + a) * 2 + 0];
  float bh = __expf(p[3]) * c_anchors[(anchor_off + a) * 2 + 1];
  float so = 1.f / (1.f + __expf(-p[4]));
  float sc = 1.f / (1.f + __expf(-p[5]));
  float bx = (sx + gx) / g;
  float by = (sy + gy) / g;
  size_t o = (size_t)n * total + box_base + r;
  boxes[o * 4 + 0] = bx - 0.5f * bw;
  boxes[o * 4 + 1] = by - 0.5f * bh;
  boxes[o * 4 + 2] = bx + 0.5f * bw;
  boxes[o * 4 + 3] = by + 0.5f * bh;
  scores[o] = so * sc;
}

// ---------------------------------------------------------------------------
// Greedy NMS, one block per image.
// ---------------------------------------------------------------------------
__global__ __launch_bounds__(256)
void nms_kernel(const float* __restrict__ boxes, float* __restrict__ scores,
                float* __restrict__ out_boxes, float* __restrict__ out_scores,
                int* __restrict__ out_cls, int* __restrict__ out_cnt, int numBoxes)
{
  int img = blockIdx.x;
  const float* bx = boxes + (size_t)img * numBoxes * 4;
  float* sc = scores + (size_t)img * numBoxes;

  __shared__ float s_val[256];
  __shared__ int   s_idx[256];
  __shared__ float best[5];
  __shared__ int   s_count;

  if (threadIdx.x == 0) s_count = 0;
  for (int i = threadIdx.x; i < 100; i += 256) {
    out_scores[img * 100 + i] = 0.f;
    out_cls[img * 100 + i] = 0;
    for (int j = 0; j < 4; ++j) out_boxes[(img * 100 + i) * 4 + j] = 0.f;
  }
  __syncthreads();

  for (int iter = 0; iter < 100; ++iter) {
    float mv = -1.f; int mi = -1;
    for (int i = threadIdx.x; i < numBoxes; i += 256) {
      float v = sc[i];
      if (v > mv) { mv = v; mi = i; }
    }
    s_val[threadIdx.x] = mv; s_idx[threadIdx.x] = mi;
    __syncthreads();
    for (int off = 128; off > 0; off >>= 1) {
      if (threadIdx.x < off && s_val[threadIdx.x + off] > s_val[threadIdx.x]) {
        s_val[threadIdx.x] = s_val[threadIdx.x + off];
        s_idx[threadIdx.x] = s_idx[threadIdx.x + off];
      }
      __syncthreads();
    }
    if (threadIdx.x == 0) {
      if (s_val[0] > 0.5f) {
        int bi = s_idx[0];
        best[0] = bx[bi * 4 + 0]; best[1] = bx[bi * 4 + 1];
        best[2] = bx[bi * 4 + 2]; best[3] = bx[bi * 4 + 3];
        best[4] = s_val[0];
        int k = s_count++;
        out_scores[img * 100 + k] = s_val[0];
        for (int j = 0; j < 4; ++j) out_boxes[(img * 100 + k) * 4 + j] = best[j];
        sc[bi] = -1.f;
      } else {
        best[4] = -1.f;
      }
    }
    __syncthreads();
    if (best[4] < 0.f) break;

    float bx0 = best[0], by0 = best[1], bx1 = best[2], by1 = best[3];
    float barea = fmaxf(bx1 - bx0, 0.f) * fmaxf(by1 - by0, 0.f);
    for (int i = threadIdx.x; i < numBoxes; i += 256) {
      float v = sc[i];
      if (v <= 0.5f) continue;
      float x0 = bx[i * 4 + 0], y0 = bx[i * 4 + 1];
      float x1 = bx[i * 4 + 2], y1 = bx[i * 4 + 3];
      float ix0 = fmaxf(x0, bx0), iy0 = fmaxf(y0, by0);
      float ix1 = fminf(x1, bx1), iy1 = fminf(y1, by1);
      float inter = fmaxf(ix1 - ix0, 0.f) * fmaxf(iy1 - iy0, 0.f);
      float area = fmaxf(x1 - x0, 0.f) * fmaxf(y1 - y0, 0.f);
      float iou = inter / (area + barea - inter + 1e-9f);
      if (iou > 0.5f) sc[i] = -1.f;
    }
    __syncthreads();
  }
  if (threadIdx.x == 0) out_cnt[img] = s_count;
}

// ---------------------------------------------------------------------------
// Host orchestration
// ---------------------------------------------------------------------------
struct DC { const float *w, *gamma, *beta, *mean, *var, *b; };
struct Cursor { void* const* in; int i; };

static DC getBN(Cursor& c) {
  DC d{};  // sorted keys: beta, gamma, mean, var, w
  d.beta  = (const float*)c.in[c.i++];
  d.gamma = (const float*)c.in[c.i++];
  d.mean  = (const float*)c.in[c.i++];
  d.var   = (const float*)c.in[c.i++];
  d.w     = (const float*)c.in[c.i++];
  return d;
}
static DC getNB(Cursor& c) {
  DC d{};  // sorted keys: b, w
  d.b = (const float*)c.in[c.i++];
  d.w = (const float*)c.in[c.i++];
  return d;
}

static void run_conv(hipStream_t st, const float* in, float* out, const DC& dc,
                     int Nb, int H, int W, int Cin, int Cout, int K, int stride,
                     const float* residual)
{
  int pad = (K == 3) ? 1 : 0;
  int OH = (stride == 1) ? H : H / 2;
  int OW = (stride == 1) ? W : W / 2;
  int M = Nb * OH * OW;
  dim3 grid((M + BM - 1) / BM, (Cout + BN - 1) / BN);
  conv_wmma_kernel<<<grid, 256, 0, st>>>(in, dc.w, dc.gamma, dc.beta, dc.mean,
                                         dc.var, dc.b, residual, out,
                                         Nb, H, W, Cin, Cout, K, stride, pad, OH, OW);
}

extern "C" void kernel_launch(void* const* d_in, const int* in_sizes, int n_in,
                              void* d_out, int out_size, void* d_ws, size_t ws_size,
                              hipStream_t stream)
{
  (void)in_sizes; (void)n_in; (void)out_size; (void)ws_size;
  const float* x = (const float*)d_in[0];

  // ---- parse params (jax pytree: dict keys sorted, lists in order) ----
  Cursor c{d_in, 1};
  DC b1[1][2], b2[2][2], b3[8][2], b4[8][2], b5[4][2];
  for (int i = 0; i < 1; ++i) { b1[i][0] = getBN(c); b1[i][1] = getBN(c); }
  for (int i = 0; i < 2; ++i) { b2[i][0] = getBN(c); b2[i][1] = getBN(c); }
  for (int i = 0; i < 8; ++i) { b3[i][0] = getBN(c); b3[i][1] = getBN(c); }
  for (int i = 0; i < 8; ++i) { b4[i][0] = getBN(c); b4[i][1] = getBN(c); }
  for (int i = 0; i < 4; ++i) { b5[i][0] = getBN(c); b5[i][1] = getBN(c); }
  DC d0 = getBN(c), d1 = getBN(c), d2 = getBN(c), d3 = getBN(c), d4 = getBN(c), d5 = getBN(c);
  DC yc1[5], yc2[5], yc3[5];
  for (int i = 0; i < 5; ++i) yc1[i] = getBN(c);
  for (int i = 0; i < 5; ++i) yc2[i] = getBN(c);
  for (int i = 0; i < 5; ++i) yc3[i] = getBN(c);
  DC yo1[2], yo2[2], yo3[2];
  yo1[0] = getBN(c); yo1[1] = getNB(c);
  yo2[0] = getBN(c); yo2[1] = getNB(c);
  yo3[0] = getBN(c); yo3[1] = getNB(c);
  DC yp2 = getBN(c), yp3 = getBN(c);

  // ---- workspace bump allocator ----
  char* wsb = (char*)d_ws;
  size_t off = 0;
  auto alloc = [&](size_t elems) -> float* {
    float* p = (float*)(wsb + off);
    off += elems * sizeof(float);
    off = (off + 255) & ~(size_t)255;
    return p;
  };
  const size_t BIGE = (size_t)2 * 416 * 416 * 32;   // largest activation
  const size_t MEDE = (size_t)2 * 52 * 52 * 512;    // largest head tensor
  float *R0 = alloc(BIGE), *R1 = alloc(BIGE), *R2 = alloc(BIGE);
  float *H1 = alloc(MEDE), *H2 = alloc(MEDE), *H3 = alloc(MEDE), *H4 = alloc(MEDE);
  float *X36 = alloc((size_t)2 * 52 * 52 * 256);
  float *X61 = alloc((size_t)2 * 26 * 26 * 512);
  float *Y1  = alloc((size_t)2 * 13 * 13 * 512);
  float *Y2  = alloc((size_t)2 * 26 * 26 * 256);
  float *O1 = alloc((size_t)2 * 13 * 13 * 18);
  float *O2 = alloc((size_t)2 * 26 * 26 * 18);
  float *O3 = alloc((size_t)2 * 52 * 52 * 18);
  float *BOXES  = alloc((size_t)2 * 10647 * 4);
  float *SCORES = alloc((size_t)2 * 10647);

  // ---- darknet-53 backbone ----
  run_conv(stream, x,  R0, d0, 2, 416, 416,  3,  32, 3, 1, nullptr);
  run_conv(stream, R0, R1, d1, 2, 416, 416, 32,  64, 3, 2, nullptr);  // 208
  float *cur = R1, *f1 = R0, *f2 = R2;
  auto res = [&](DC* pair, int Hc, int C) {
    run_conv(stream, cur, f1, pair[0], 2, Hc, Hc, C, C / 2, 1, 1, nullptr);
    run_conv(stream, f1,  f2, pair[1], 2, Hc, Hc, C / 2, C, 3, 1, cur);
    float* t = cur; cur = f2; f2 = t;
  };
  auto down = [&](const DC& dc, int Hc, int Cin, int Cout) {
    run_conv(stream, cur, f1, dc, 2, Hc, Hc, Cin, Cout, 3, 2, nullptr);
    float* t = cur; cur = f1; f1 = t;
  };
  for (int i = 0; i < 1; ++i) res(b1[i], 208, 64);
  down(d2, 208, 64, 128);
  for (int i = 0; i < 2; ++i) res(b2[i], 104, 128);
  down(d3, 104, 128, 256);
  for (int i = 0; i < 8; ++i) res(b3[i], 52, 256);
  hipMemcpyAsync(X36, cur, (size_t)2 * 52 * 52 * 256 * sizeof(float),
                 hipMemcpyDeviceToDevice, stream);
  down(d4, 52, 256, 512);
  for (int i = 0; i < 8; ++i) res(b4[i], 26, 512);
  hipMemcpyAsync(X61, cur, (size_t)2 * 26 * 26 * 512 * sizeof(float),
                 hipMemcpyDeviceToDevice, stream);
  down(d5, 26, 512, 1024);
  for (int i = 0; i < 4; ++i) res(b5[i], 13, 1024);

  // ---- YoloConv 5-conv stacks ----
  auto run_yc = [&](const float* xin, DC* convs, int Hc, int cin0, int f,
                    float* s1, float* s2) -> float* {
    const float* src = xin; float* dst = s1; float* other = s2;
    int cins[5]  = {cin0, f, 2 * f, f, 2 * f};
    int couts[5] = {f, 2 * f, f, 2 * f, f};
    int ks[5]    = {1, 3, 1, 3, 1};
    for (int i = 0; i < 5; ++i) {
      run_conv(stream, src, dst, convs[i], 2, Hc, Hc, cins[i], couts[i], ks[i], 1, nullptr);
      src = dst; float* t = dst; dst = other; other = t;
    }
    return (float*)src;
  };

  // head 1 (13x13)
  float* y1 = run_yc(cur, yc1, 13, 1024, 512, H1, H2);
  hipMemcpyAsync(Y1, y1, (size_t)2 * 13 * 13 * 512 * sizeof(float),
                 hipMemcpyDeviceToDevice, stream);
  run_conv(stream, Y1, H3, yo1[0], 2, 13, 13, 512, 1024, 3, 1, nullptr);
  run_conv(stream, H3, O1, yo1[1], 2, 13, 13, 1024, 18, 1, 1, nullptr);

  // head 2 (26x26): upsample -> 1x1 conv -> concat(x61) -> yc2
  {
    size_t tot = (size_t)2 * 26 * 26 * 512;
    upsample2x_kernel<<<(unsigned)((tot + 255) / 256), 256, 0, stream>>>(Y1, H3, 2, 13, 13, 512);
    run_conv(stream, H3, H4, yp2, 2, 26, 26, 512, 256, 1, 1, nullptr);
    size_t ct = (size_t)2 * 26 * 26 * 768;
    concat_ch_kernel<<<(unsigned)((ct + 255) / 256), 256, 0, stream>>>(H4, X61, H1, 2, 26, 26, 256, 512);
  }
  float* y2 = run_yc(H1, yc2, 26, 768, 256, H2, H3);
  hipMemcpyAsync(Y2, y2, (size_t)2 * 26 * 26 * 256 * sizeof(float),
                 hipMemcpyDeviceToDevice, stream);
  run_conv(stream, Y2, H4, yo2[0], 2, 26, 26, 256, 512, 3, 1, nullptr);
  run_conv(stream, H4, O2, yo2[1], 2, 26, 26, 512, 18, 1, 1, nullptr);

  // head 3 (52x52)
  {
    size_t tot = (size_t)2 * 52 * 52 * 256;
    upsample2x_kernel<<<(unsigned)((tot + 255) / 256), 256, 0, stream>>>(Y2, H3, 2, 26, 26, 256);
    run_conv(stream, H3, H4, yp3, 2, 52, 52, 256, 128, 1, 1, nullptr);
    size_t ct = (size_t)2 * 52 * 52 * 384;
    concat_ch_kernel<<<(unsigned)((ct + 255) / 256), 256, 0, stream>>>(H4, X36, H1, 2, 52, 52, 128, 256);
  }
  float* y3 = run_yc(H1, yc3, 52, 384, 128, H2, H3);
  run_conv(stream, y3, H4, yo3[0], 2, 52, 52, 128, 256, 3, 1, nullptr);
  run_conv(stream, H4, O3, yo3[1], 2, 52, 52, 256, 18, 1, 1, nullptr);

  // ---- decode all three scales into (2,10647,4) boxes / (2,10647) scores ----
  const int TOTAL = 10647;
  {
    int n1 = 2 * 13 * 13 * 3;
    decode_kernel<<<(n1 + 255) / 256, 256, 0, stream>>>(O1, BOXES, SCORES, 2, 13, 6, 0, TOTAL);
    int n2 = 2 * 26 * 26 * 3;
    decode_kernel<<<(n2 + 255) / 256, 256, 0, stream>>>(O2, BOXES, SCORES, 2, 26, 3, 507, TOTAL);
    int n3 = 2 * 52 * 52 * 3;
    decode_kernel<<<(n3 + 255) / 256, 256, 0, stream>>>(O3, BOXES, SCORES, 2, 52, 0, 2535, TOTAL);
  }

  // ---- NMS -> output tuple (boxes[2,100,4], scores[2,100], cls[2,100], cnt[2]) ----
  float* out = (float*)d_out;
  float* out_boxes  = out;                 // 800 floats
  float* out_scores = out + 800;           // 200 floats
  int*   out_cls    = (int*)(out + 1000);  // 200 ints
  int*   out_cnt    = (int*)(out + 1200);  // 2 ints
  nms_kernel<<<2, 256, 0, stream>>>(BOXES, SCORES, out_boxes, out_scores,
                                    out_cls, out_cnt, TOTAL);
}
